// FrameSimilarity_11192684773696
// MI455X (gfx1250) — matrix operand order repaired and checked by reading, not compiled, not run on hardware
//
#include <hip/hip_runtime.h>

// MI455X / gfx1250, wave32. WMMA f16 16x16x32, f32 accumulate.
// Stage 1 uses async global->LDS (ASYNCcnt) staging when available.

typedef __attribute__((ext_vector_type(16))) _Float16 v16h;
typedef __attribute__((ext_vector_type(8)))  _Float16 v8h;
typedef __attribute__((ext_vector_type(8)))  float    v8f;
typedef __attribute__((ext_vector_type(4)))  float    v4f;
typedef __attribute__((ext_vector_type(4)))  int      v4i;

#define B_      4
#define C_IN    2048     // concat channels
#define T_      512
#define D_      128
#define LOOKUP_ 101
#define HALFW_  50

// A-fragment K index for element i of v16h, given lane-half h (lane>>4):
// lanes 0-15: K = 0..7,16..23 ; lanes 16-31: K = 8..15,24..31
__device__ __forceinline__ int a_kmap(int i, int h) {
    return i + 8 * ((i >> 3) + h);
}

// ---------------------------------------------------------------------------
// Kernel 1: 7x7 spatial mean of x1,x2 -> feats[b][c][t] (f16)
// Bandwidth-dominant stage (822 MB @ 23.3 TB/s ~= 35 us floor).
// Block stages its contiguous 256x49-float region into LDS with coalesced
// 16B async global->LDS transfers (512 B / wave instruction, 4 full
// cachelines), then each thread reduces 49 floats from LDS (stride 49
// dwords: 49 coprime to 64 banks -> conflict-free).
// ---------------------------------------------------------------------------
__global__ __launch_bounds__(256) void mean_reduce_kernel(
    const float* __restrict__ x1, const float* __restrict__ x2,
    _Float16* __restrict__ feats) {
    __shared__ float tile[256 * 49];                  // 50176 B
    const int tid  = threadIdx.x;
    const int out0 = blockIdx.x * 256;                // first output of block
    // 256 outputs share one (b,c) row: t0 in {0,256}
    const int t0 = out0 & (T_ - 1);
    const int c  = (out0 >> 9) & (C_IN - 1);
    const int b  = out0 >> 20;                        // 2048*512 = 2^20
    const float* src = (c < 1024) ? x1 : x2;
    const int cc = c & 1023;
    const float* gbase = src + ((size_t)((b * 1024 + cc) * T_ + t0)) * 49;
    // region byte offset is a multiple of 256*196 = 50176 -> 16B aligned

#if __has_builtin(__builtin_amdgcn_global_load_async_to_lds_b128)
    for (int i = tid; i < 3136; i += 256) {           // 3136 x 16B = 50176 B
        __builtin_amdgcn_global_load_async_to_lds_b128(
            (__attribute__((address_space(1))) v4i*)(gbase + i * 4),
            (__attribute__((address_space(3))) v4i*)(tile + i * 4),
            0, 0);
    }
  #if __has_builtin(__builtin_amdgcn_s_wait_asynccnt)
    __builtin_amdgcn_s_wait_asynccnt(0);
  #else
    asm volatile("s_wait_asynccnt 0x0" ::: "memory");
  #endif
#else
    for (int i = tid; i < 3136; i += 256) {
        const v4f v = __builtin_nontemporal_load((const v4f*)gbase + i);
        *(v4f*)(tile + i * 4) = v;
    }
#endif
    __syncthreads();

    const float* p = tile + tid * 49;
    float s = 0.0f;
#pragma unroll
    for (int j = 0; j < 49; ++j) s += p[j];
    feats[out0 + tid] = (_Float16)(s * (1.0f / 49.0f));
}

// ---------------------------------------------------------------------------
// Kernel 2: projection GEMM (M=T tile of 16, N=D=128, K=C=2048) + L2 normalize
// One block per (b, t-tile); 8 waves, wave w owns d-tile [16w,16w+16).
// Row sum-of-squares reduced across waves via LDS ds_add_f32.
// ---------------------------------------------------------------------------
__global__ __launch_bounds__(256) void proj_norm_kernel(
    const _Float16* __restrict__ feats, const float* __restrict__ Wp,
    _Float16* __restrict__ xn) {
    const int tid  = threadIdx.x;
    const int wave = tid >> 5;
    const int lane = tid & 31;
    const int h    = lane >> 4;
    const int lr   = lane & 15;
    const int t0   = (blockIdx.x & 31) << 4;
    const int b    = blockIdx.x >> 5;
    const int n    = (wave << 4) + lr;        // output column d

    __shared__ float ssq[16];
    if (tid < 16) ssq[tid] = 0.0f;
    __syncthreads();

    const _Float16* fb = feats + b * (C_IN * T_);   // [c][t]
    v8f acc = {};
    for (int k0 = 0; k0 < C_IN; k0 += 32) {
        v16h a, bm;
#pragma unroll
        for (int i = 0; i < 16; ++i) {
            const int K = k0 + a_kmap(i, h);
            a[i] = fb[K * T_ + (t0 + lr)];
        }
        const float* wrow = Wp + n * C_IN + k0; // K-step base; table-order K via a_kmap
#pragma unroll
        for (int i = 0; i < 16; ++i) bm[i] = (_Float16)wrow[a_kmap(i, h)];
        acc = __builtin_amdgcn_wmma_f32_16x16x32_f16(
            false, a, false, bm, (short)0, acc, false, false);
    }

#pragma unroll
    for (int r = 0; r < 8; ++r)
        atomicAdd(&ssq[r + (h << 3)], acc[r] * acc[r]);   // ds_add_f32
    __syncthreads();

#pragma unroll
    for (int r = 0; r < 8; ++r) {
        const int m = r + (h << 3);
        const float inv = 1.0f / fmaxf(sqrtf(ssq[m]), 1e-12f);
        xn[(b * T_ + (t0 + m)) * D_ + n] = (_Float16)(acc[r] * inv);
    }
}

// ---------------------------------------------------------------------------
// Kernel 3: banded cosine sims (9 s-tiles per t-tile via WMMA) -> LDS window
//           -> FC [101->128] via K-padded WMMA + bias + ReLU
// One block per (b, t-tile); 8 waves.
// ---------------------------------------------------------------------------
__global__ __launch_bounds__(256) void band_fc_kernel(
    const _Float16* __restrict__ xn, const float* __restrict__ Wf,
    const float* __restrict__ bf, float* __restrict__ out) {
    const int tid  = threadIdx.x;
    const int wave = tid >> 5;
    const int lane = tid & 31;
    const int h    = lane >> 4;
    const int lr   = lane & 15;
    const int t0   = (blockIdx.x & 31) << 4;
    const int b    = blockIdx.x >> 5;

    __shared__ float win[16][112];            // 16 rows x (101 padded), zeroed
    for (int i = tid; i < 16 * 112; i += 256) ((float*)win)[i] = 0.0f;
    __syncthreads();

    const _Float16* xb = xn + b * (T_ * D_);  // [t][d]

    // --- banded sims: s-tiles at s0 = t0-64+16j, j=0..8 cover s in [t-50,t+50]
    for (int j = wave; j < 9; j += 8) {
        const int s0 = t0 - 64 + (j << 4);
        const int s  = s0 + lr;               // this lane's column
        v8f acc = {};
#pragma unroll
        for (int k0 = 0; k0 < D_; k0 += 32) {
            v16h a, bm;
            const _Float16* arow = xb + (t0 + lr) * D_;
            const v8h* ar = (const v8h*)(arow + k0 + (h << 3));
            const v8h a0 = ar[0];             // K = k0+8h   .. +7
            const v8h a1 = ar[2];             // K = k0+16+8h.. +7
#pragma unroll
            for (int i = 0; i < 8; ++i) { a[i] = a0[i]; a[8 + i] = a1[i]; }
            if (s >= 0 && s < T_) {
                const v8h* br = (const v8h*)(xb + s * D_ + k0 + (h << 3));
                const v8h b0 = br[0], b1 = br[2];
#pragma unroll
                for (int i = 0; i < 8; ++i) { bm[i] = b0[i]; bm[8 + i] = b1[i]; }
            } else {
#pragma unroll
                for (int i = 0; i < 16; ++i) bm[i] = (_Float16)0.0f;
            }
            acc = __builtin_amdgcn_wmma_f32_16x16x32_f16(
                false, a, false, bm, (short)0, acc, false, false);
        }
#pragma unroll
        for (int r = 0; r < 8; ++r) {
            const int m = r + (h << 3);
            const int l = s - (t0 + m) + HALFW_;
            if (s >= 0 && s < T_ && l >= 0 && l < LOOKUP_) win[m][l] = acc[r];
        }
    }
    __syncthreads();

    // --- FC: M=16 t-rows, N=128 outputs (wave w owns o-tile), K=101 pad->128
    const int o = (wave << 4) + lr;
    v8f acc = {};
#pragma unroll
    for (int k0 = 0; k0 < 128; k0 += 32) {
        v16h a, bm;
#pragma unroll
        for (int i = 0; i < 16; ++i) {
            const int K = k0 + a_kmap(i, h);
            a[i] = (K < LOOKUP_) ? (_Float16)win[lr][K] : (_Float16)0.0f;
        }
#pragma unroll
        for (int i = 0; i < 16; ++i) {
            const int K = k0 + a_kmap(i, h);
            bm[i] = (K < LOOKUP_) ? (_Float16)Wf[o * LOOKUP_ + K]
                                  : (_Float16)0.0f;
        }
        acc = __builtin_amdgcn_wmma_f32_16x16x32_f16(
            false, a, false, bm, (short)0, acc, false, false);
    }
    const float bias = bf[o];
#pragma unroll
    for (int r = 0; r < 8; ++r) {
        const int t = t0 + r + (h << 3);
        const float v = acc[r] + bias;
        out[(b * T_ + t) * D_ + o] = v > 0.0f ? v : 0.0f;
    }
}

// ---------------------------------------------------------------------------
extern "C" void kernel_launch(void* const* d_in, const int* in_sizes, int n_in,
                              void* d_out, int out_size, void* d_ws, size_t ws_size,
                              hipStream_t stream) {
    const float* x1 = (const float*)d_in[0];
    const float* x2 = (const float*)d_in[1];
    const float* Wp = (const float*)d_in[2];
    const float* Wf = (const float*)d_in[3];
    const float* bf = (const float*)d_in[4];
    float* out = (float*)d_out;

    // workspace layout (needs ~8.9 MB)
    _Float16* feats = (_Float16*)d_ws;                               // 8 MB
    _Float16* xn    = (_Float16*)((char*)d_ws +
                                  (size_t)B_ * C_IN * T_ * sizeof(_Float16));

    // stage 1: 4*2048*512 outputs, one thread each (exact grid)
    mean_reduce_kernel<<<(B_ * C_IN * T_) / 256, 256, 0, stream>>>(x1, x2, feats);
    // stages 2/3: one block per (b, 16-row t-tile) = 4*32 = 128 blocks
    proj_norm_kernel<<<B_ * (T_ / 16), 256, 0, stream>>>(feats, Wp, xn);
    band_fc_kernel<<<B_ * (T_ / 16), 256, 0, stream>>>(xn, Wf, bf, out);
}
